// MultiHeadAttention_73624329388209
// MI455X (gfx1250) — matrix operand order repaired and checked by reading, not compiled, not run on hardware
//
#include <hip/hip_runtime.h>

#define DEV_INLINE __device__ __forceinline__

typedef __attribute__((ext_vector_type(16))) __bf16 v16bf;
typedef __attribute__((ext_vector_type(8)))  float  v8f;

union Frag  { v16bf v; uint4 u[2]; unsigned w[8]; };
union Pack8 { uint4 q[2]; unsigned short s[16]; };

constexpr int B_  = 4;
constexpr int S_  = 2048;
constexpr int D_  = 1024;
constexpr int H_  = 16;
constexpr int DK_ = 64;
constexpr int BS_ = B_ * S_;   // 8192 rows

// scale for Q projection: (1/sqrt(dk)) * log2(e)  -> flash loop works in
// log2 domain; raw v_exp_f32 (exp2) needs no pre-multiply at all.
#define QSCALE (0.125f * 1.44269504088896340736f)

// raw hardware exp2: single v_exp_f32, no denormal-range fixup (flushes to 0,
// exactly what flash softmax wants for vanishing probabilities)
DEV_INLINE float fast_exp2(float x) { return __builtin_amdgcn_exp2f(x); }

DEV_INLINE unsigned short bf_bits(float f) {
    __bf16 h = (__bf16)f;
    return __builtin_bit_cast(unsigned short, h);
}
DEV_INLINE unsigned pack_bf16(float lo, float hi) {
    return ((unsigned)bf_bits(hi) << 16) | bf_bits(lo);
}

// ---------------------------------------------------------------------------
// Stage 0a: plain f32 -> bf16 cast (activations, natural layout)
// ---------------------------------------------------------------------------
__global__ void cast_bf16_kernel(const float* __restrict__ in,
                                 unsigned short* __restrict__ out, int n) {
    int i = blockIdx.x * blockDim.x + threadIdx.x;
    if (i < n) out[i] = bf_bits(in[i]);
}

// ---------------------------------------------------------------------------
// Stage 0b: tiled transpose-cast for weights: Wt[n][k] = bf16(W[k][n])
// ---------------------------------------------------------------------------
__global__ __launch_bounds__(256) void castT_bf16_kernel(
    const float* __restrict__ in, unsigned short* __restrict__ out) {
    __shared__ unsigned short T[64 * 65];
    const int kb = blockIdx.x * 64;
    const int nb = blockIdx.y * 64;
    const int t  = threadIdx.x;
    const int row = t >> 2;            // 0..63
    const int c16 = (t & 3) * 16;      // 0,16,32,48

    const float4* g = (const float4*)&in[(size_t)(kb + row) * D_ + nb + c16];
#pragma unroll
    for (int j = 0; j < 4; ++j) {
        float4 v = g[j];
        int c = c16 + j * 4;
        T[(c + 0) * 65 + row] = bf_bits(v.x);
        T[(c + 1) * 65 + row] = bf_bits(v.y);
        T[(c + 2) * 65 + row] = bf_bits(v.z);
        T[(c + 3) * 65 + row] = bf_bits(v.w);
    }
    __syncthreads();

    Pack8 ob[2];
#pragma unroll
    for (int j = 0; j < 16; ++j) ob[j >> 3].s[j & 7] = T[row * 65 + c16 + j];
    unsigned short* o = &out[(size_t)(nb + row) * D_ + kb + c16];
    *(uint4*)&o[0] = ob[0].q[0];
    *(uint4*)&o[8] = ob[1].q[0];
}

// ---------------------------------------------------------------------------
// Transposed WMMA GEMM: computes C^T = W^T * X^T  (i.e. C = X*W + bias)
// Software-pipelined: chunk kc+1 prefetched into registers (clamped index,
// no loop peel) while chunk kc is consumed by WMMA.
// MODE 0: bf16 out, head layout [b][h][s][dk], scaled     (Q, K projections)
// MODE 2: bf16 out, transposed head layout [bh][dk][s]    (V projection)
// MODE 1: f32 out, row-major [8192][1024]                 (final projection)
// ---------------------------------------------------------------------------
template <int MODE>
__global__ __launch_bounds__(256) void gemmT_bf16_kernel(
    const unsigned short* __restrict__ Wt,   // [1024][1024] bf16, [n][k]
    const unsigned short* __restrict__ X,    // [8192][1024] bf16, [m][k]
    const float* __restrict__ bias,          // [1024], indexed by n
    void* __restrict__ out, float scale)
{
    __shared__ unsigned short At[128 * 32];  // [n][k]
    __shared__ unsigned short Xt[64 * 32];   // [m][k]

    const int tid   = threadIdx.x;
    const int lane  = tid & 31;
    const int wave  = tid >> 5;
    const int col16 = lane & 15;
    const int hi    = lane >> 4;
    const int Nbase = blockIdx.x * 128;      // out-col tile (M dim)
    const int Mbase = blockIdx.y * 64;       // row tile (N dim)

    // per-thread staging source pointers (advance by 32 halfs per K-chunk)
    const unsigned short* gA = &Wt[(size_t)(Nbase + (tid >> 1)) * D_ + (tid & 1) * 16];
    const unsigned short* gX = &X [(size_t)(Mbase + (tid >> 2)) * D_ + (tid & 3) * 8];
    uint4* lA = (uint4*)&At[(tid >> 1) * 32 + (tid & 1) * 16];
    uint4* lX = (uint4*)&Xt[(tid >> 2) * 32 + (tid & 3) * 8];

    v8f acc[4] = {};
    uint4 rA0, rA1, rX;

    rA0 = ((const uint4*)gA)[0];
    rA1 = ((const uint4*)gA)[1];
    rX  = *(const uint4*)gX;

    constexpr int NC = D_ / 32;
    for (int kc = 0; kc < NC; ++kc) {
        lA[0] = rA0;
        lA[1] = rA1;
        *lX   = rX;
        __syncthreads();

        {   // prefetch next chunk (clamped: last iter redundantly reloads)
            int nx = (kc + 1 < NC) ? (kc + 1) : kc;
            const unsigned short* nA = gA + nx * 32;
            const unsigned short* nX = gX + nx * 32;
            rA0 = ((const uint4*)nA)[0];
            rA1 = ((const uint4*)nA)[1];
            rX  = *(const uint4*)nX;
        }

        Frag fa;   // A fragment: row n = wave*16+col16
        {
            int row = wave * 16 + col16;
            fa.u[0] = *(const uint4*)&At[row * 32 + hi * 8];
            fa.u[1] = *(const uint4*)&At[row * 32 + 16 + hi * 8];
        }
#pragma unroll
        for (int mt = 0; mt < 4; ++mt) {     // B fragments: col m = mt*16+col16
            Frag fb;
            int m = mt * 16 + col16;
            fb.u[0] = *(const uint4*)&Xt[m * 32 + hi * 8];
            fb.u[1] = *(const uint4*)&Xt[m * 32 + 16 + hi * 8];
            acc[mt] = __builtin_amdgcn_wmma_f32_16x16x32_bf16(
                false, fa.v, false, fb.v, (short)0, acc[mt], false, false);
        }
        __syncthreads();
    }

    // epilogue: lane holds out-cols n0..n0+7 (contiguous) at row m
    const int n0 = Nbase + wave * 16 + hi * 8;
    float bv[8];
#pragma unroll
    for (int r = 0; r < 8; ++r) bv[r] = bias[n0 + r];

#pragma unroll
    for (int mt = 0; mt < 4; ++mt) {
        const int m = Mbase + mt * 16 + col16;
        if (MODE == 0) {        // bf16, [b][h][s][dk], scaled, packed 16B store
            int b = m / S_, s = m % S_;
            int h = n0 / DK_, dk0 = n0 % DK_;
            Pack8 p;
#pragma unroll
            for (int r = 0; r < 8; ++r)
                p.s[r] = bf_bits((acc[mt][r] + bv[r]) * scale);
            *(uint4*)&((unsigned short*)out)[(((size_t)b * H_ + h) * S_ + s) * DK_ + dk0] = p.q[0];
        } else if (MODE == 2) { // bf16, [bh][dk][s] (V transposed)
            int b = m / S_, s = m % S_;
            int h = n0 / DK_, dk0 = n0 % DK_;
            size_t base = ((size_t)(b * H_ + h) * DK_ + dk0) * S_ + s;
#pragma unroll
            for (int r = 0; r < 8; ++r)
                ((unsigned short*)out)[base + (size_t)r * S_] = bf_bits(acc[mt][r] + bv[r]);
        } else {                // f32 row-major, two 16B stores
            float4 v0 = { acc[mt][0] + bv[0], acc[mt][1] + bv[1],
                          acc[mt][2] + bv[2], acc[mt][3] + bv[3] };
            float4 v1 = { acc[mt][4] + bv[4], acc[mt][5] + bv[5],
                          acc[mt][6] + bv[6], acc[mt][7] + bv[7] };
            float* o = &((float*)out)[(size_t)m * D_ + n0];
            *(float4*)&o[0] = v0;
            *(float4*)&o[4] = v1;
        }
    }
}

// ---------------------------------------------------------------------------
// Flash attention, transposed orientation: S^T = K * Q^T, O^T = V^T * P^T.
// grid = (S/128, B*H), block = 256 (8 waves); wave owns 16 queries (N dim).
// Softmax state is a per-lane scalar; scores arrive pre-multiplied by
// (1/sqrt(dk))*log2(e), so the whole flash update is raw v_exp_f32.
// K/V chunk staging is software-pipelined through registers.
//   Qp : [bh][s][64]  bf16, pre-scaled      Kp : [bh][s][64]  bf16
//   Vp : [bh][64][s]  bf16 (transposed)     O  : [8192][1024] bf16 head-merged
// ---------------------------------------------------------------------------
__global__ __launch_bounds__(256) void attn_kernel(
    const unsigned short* __restrict__ Qp,
    const unsigned short* __restrict__ Kp,
    const unsigned short* __restrict__ Vp,
    unsigned short* __restrict__ O)
{
    __shared__ unsigned short Kt[32 * 64];   // [key][dk]
    __shared__ unsigned short Vt[64 * 32];   // [dk][key]

    const int tid   = threadIdx.x;
    const int lane  = tid & 31;
    const int wave  = tid >> 5;
    const int col16 = lane & 15;
    const int hi    = lane >> 4;
    const int bh    = blockIdx.y;
    const int qbase = blockIdx.x * 128 + wave * 16;
    const int q     = qbase + col16;

    // Q^T B-fragments, resident in registers: fq[c] covers dk = c*32..c*32+31
    Frag fq[2];
    {
        const unsigned short* qrow = &Qp[((size_t)bh * S_ + q) * DK_];
#pragma unroll
        for (int c = 0; c < 2; ++c) {
            fq[c].u[0] = *(const uint4*)&qrow[c * 32 + hi * 8];
            fq[c].u[1] = *(const uint4*)&qrow[c * 32 + 16 + hi * 8];
        }
    }

    // staging pointers: K advances 32*64 halfs/chunk, V advances 32 halfs/chunk
    const unsigned short* gK = &Kp[((size_t)bh * S_ + (tid >> 3)) * DK_ + (tid & 7) * 8];
    const unsigned short* gV = &Vp[((size_t)bh * DK_ + (tid >> 2)) * S_ + (tid & 3) * 8];
    uint4* lK = (uint4*)&Kt[(tid >> 3) * 64 + (tid & 7) * 8];
    uint4* lV = (uint4*)&Vt[(tid >> 2) * 32 + (tid & 3) * 8];

    v8f acc[4] = {};            // O^T: acc[mt], dk = mt*16 + r + hi*8, col q
    float mval = -1e30f, lval = 0.f;

    uint4 rK = *(const uint4*)gK;
    uint4 rV = *(const uint4*)gV;

    constexpr int NC = S_ / 32;
    for (int kc = 0; kc < NC; ++kc) {
        *lK = rK;
        *lV = rV;
        __syncthreads();

        {   // prefetch next K/V chunk during compute (clamped, no peel)
            int nx = (kc + 1 < NC) ? (kc + 1) : kc;
            rK = *(const uint4*)(gK + (size_t)nx * 32 * DK_);
            rV = *(const uint4*)(gV + (size_t)nx * 32);
        }

        // S^T tiles: rows = keys (2 tiles of 16), cols = q
        v8f s0 = {}, s1 = {};
#pragma unroll
        for (int c = 0; c < 2; ++c) {
            Frag fk0, fk1;
            int key0 = col16, key1 = 16 + col16;
            fk0.u[0] = *(const uint4*)&Kt[key0 * 64 + c * 32 + hi * 8];
            fk0.u[1] = *(const uint4*)&Kt[key0 * 64 + c * 32 + 16 + hi * 8];
            s0 = __builtin_amdgcn_wmma_f32_16x16x32_bf16(
                false, fk0.v, false, fq[c].v, (short)0, s0, false, false);
            fk1.u[0] = *(const uint4*)&Kt[key1 * 64 + c * 32 + hi * 8];
            fk1.u[1] = *(const uint4*)&Kt[key1 * 64 + c * 32 + 16 + hi * 8];
            s1 = __builtin_amdgcn_wmma_f32_16x16x32_bf16(
                false, fk1.v, false, fq[c].v, (short)0, s1, false, false);
        }

        // online softmax in log2 domain (raw v_exp_f32, no fixup code)
        float mx = fmaxf(s0[0], s1[0]);
#pragma unroll
        for (int r = 1; r < 8; ++r) mx = fmaxf(mx, fmaxf(s0[r], s1[r]));
        mx = fmaxf(mx, __shfl_xor(mx, 16, 32));
        float mn    = fmaxf(mval, mx);
        float alpha = fast_exp2(mval - mn);
        float ps    = 0.f;
#pragma unroll
        for (int r = 0; r < 8; ++r) {
            s0[r] = fast_exp2(s0[r] - mn);
            s1[r] = fast_exp2(s1[r] - mn);
            ps += s0[r] + s1[r];
        }
        ps += __shfl_xor(ps, 16, 32);
        lval = lval * alpha + ps;
        mval = mn;
#pragma unroll
        for (int mt = 0; mt < 4; ++mt)
#pragma unroll
            for (int r = 0; r < 8; ++r) acc[mt][r] *= alpha;

        // build P^T B-fragment in registers: pack bf16 pairs + 4-word exchange
        Frag fp;
        {
            unsigned p0w[4], p1w[4], xw[4];
#pragma unroll
            for (int j = 0; j < 4; ++j) {
                p0w[j] = pack_bf16(s0[2 * j], s0[2 * j + 1]);
                p1w[j] = pack_bf16(s1[2 * j], s1[2 * j + 1]);
                xw[j]  = hi ? p0w[j] : p1w[j];
                xw[j]  = (unsigned)__shfl_xor((int)xw[j], 16, 32);
            }
#pragma unroll
            for (int j = 0; j < 4; ++j) {
                fp.w[j]     = hi ? xw[j] : p0w[j];
                fp.w[4 + j] = hi ? p1w[j] : xw[j];
            }
        }

        // O^T += V^T * P^T
#pragma unroll
        for (int mt = 0; mt < 4; ++mt) {
            Frag fv;
            int dk = mt * 16 + col16;
            fv.u[0] = *(const uint4*)&Vt[dk * 32 + hi * 16];
            fv.u[1] = *(const uint4*)&Vt[dk * 32 + hi * 16 + 8];
            acc[mt] = __builtin_amdgcn_wmma_f32_16x16x32_bf16(
                false, fv.v, false, fp.v, (short)0, acc[mt], false, false);
        }
        __syncthreads();
    }

    // epilogue: lane holds one output row (b*S+q), 8 consecutive cols per mt
    const int b = bh / H_, h = bh % H_;
    const float rl = 1.f / lval;
    unsigned short* orow = &O[(size_t)(b * S_ + q) * D_ + h * DK_ + hi * 8];
#pragma unroll
    for (int mt = 0; mt < 4; ++mt) {
        Pack8 p;
#pragma unroll
        for (int r = 0; r < 8; ++r) p.s[r] = bf_bits(acc[mt][r] * rl);
        *(uint4*)&orow[mt * 16] = p.q[0];
    }
}

// ---------------------------------------------------------------------------
// Host-side orchestration
// ---------------------------------------------------------------------------
extern "C" void kernel_launch(void* const* d_in, const int* in_sizes, int n_in,
                              void* d_out, int out_size, void* d_ws, size_t ws_size,
                              hipStream_t stream) {
    const float* query = (const float*)d_in[0];
    const float* key   = (const float*)d_in[1];
    const float* value = (const float*)d_in[2];
    const float* Wq = (const float*)d_in[3];
    const float* bq = (const float*)d_in[4];
    const float* Wk = (const float*)d_in[5];
    const float* bk = (const float*)d_in[6];
    const float* Wv = (const float*)d_in[7];
    const float* bv = (const float*)d_in[8];
    const float* Wo = (const float*)d_in[9];
    const float* bo = (const float*)d_in[10];

    const size_t NX = (size_t)BS_ * D_;
    const size_t NW = (size_t)D_ * D_;

    char* ws = (char*)d_ws;
    size_t off = 0;
    auto alloc = [&](size_t bytes) -> void* {
        void* p = ws + off;
        off += (bytes + 255) & ~(size_t)255;
        return p;
    };
    unsigned short* qb  = (unsigned short*)alloc(NX * 2);
    unsigned short* kb_ = (unsigned short*)alloc(NX * 2);
    unsigned short* vb  = (unsigned short*)alloc(NX * 2);
    unsigned short* wqt = (unsigned short*)alloc(NW * 2);
    unsigned short* wkt = (unsigned short*)alloc(NW * 2);
    unsigned short* wvt = (unsigned short*)alloc(NW * 2);
    unsigned short* wot = (unsigned short*)alloc(NW * 2);
    unsigned short* Qp  = (unsigned short*)alloc(NX * 2);  // [bh][s][64], pre-scaled
    unsigned short* Kp  = (unsigned short*)alloc(NX * 2);  // [bh][s][64]
    unsigned short* Vp  = (unsigned short*)alloc(NX * 2);  // [bh][64][s]
    unsigned short* Ao  = (unsigned short*)alloc(NX * 2);  // head-merged attn out

    // Stage 0: casts (activations plain, weights transposed+cast)
    const int CT = 256;
    cast_bf16_kernel<<<(int)(NX / CT), CT, 0, stream>>>(query, qb,  (int)NX);
    cast_bf16_kernel<<<(int)(NX / CT), CT, 0, stream>>>(key,   kb_, (int)NX);
    cast_bf16_kernel<<<(int)(NX / CT), CT, 0, stream>>>(value, vb,  (int)NX);
    dim3 tgrid(D_ / 64, D_ / 64);
    castT_bf16_kernel<<<tgrid, 256, 0, stream>>>(Wq, wqt);
    castT_bf16_kernel<<<tgrid, 256, 0, stream>>>(Wk, wkt);
    castT_bf16_kernel<<<tgrid, 256, 0, stream>>>(Wv, wvt);
    castT_bf16_kernel<<<tgrid, 256, 0, stream>>>(Wo, wot);

    // Stage 1: projections (Q pre-scaled by log2e/sqrt(dk); V stored transposed)
    dim3 ggrid(D_ / 128, BS_ / 64);
    gemmT_bf16_kernel<0><<<ggrid, 256, 0, stream>>>(wqt, qb,  bq, Qp, QSCALE);
    gemmT_bf16_kernel<0><<<ggrid, 256, 0, stream>>>(wkt, kb_, bk, Kp, 1.0f);
    gemmT_bf16_kernel<2><<<ggrid, 256, 0, stream>>>(wvt, vb,  bv, Vp, 1.0f);

    // Stage 2: flash attention
    dim3 agrid(S_ / 128, B_ * H_);
    attn_kernel<<<agrid, 256, 0, stream>>>(Qp, Kp, Vp, Ao);

    // Stage 3: output projection -> f32
    gemmT_bf16_kernel<1><<<ggrid, 256, 0, stream>>>(wot, Ao, bo, (float*)d_out, 1.0f);
}
